// MMDiTBlock_22694607192351
// MI455X (gfx1250) — matrix operand order repaired and checked by reading, not compiled, not run on hardware
//
#include <hip/hip_runtime.h>

// ---------------------------------------------------------------------------
// Types / helpers
// ---------------------------------------------------------------------------
typedef __attribute__((ext_vector_type(16))) __bf16 v16bf;
typedef __attribute__((ext_vector_type(8)))  float  v8f;

struct U4 { unsigned int x, y, z, w; };          // POD 16-byte chunk
union FragBF { v16bf v; U4 q[2]; };              // 32B WMMA operand fragment

__device__ __forceinline__ unsigned short f2bf(float f) {
  unsigned int u = __float_as_uint(f);
  unsigned int r = u + 0x7FFFu + ((u >> 16) & 1u);   // round-to-nearest-even
  return (unsigned short)(r >> 16);
}
__device__ __forceinline__ float bf2f(unsigned short s) {
  return __uint_as_float(((unsigned int)s) << 16);
}

// ---------------------------------------------------------------------------
// Weight cast + transpose: src (K x N) fp32 -> dst (N x K) bf16.
// Tiled 32x32 through LDS, coalesced on both sides.
// ---------------------------------------------------------------------------
__global__ __launch_bounds__(256) void cast_transpose_bf16(
    const float* __restrict__ src, unsigned short* __restrict__ dst,
    int K, int N) {
  __shared__ float tile[32][33];
  const int k0 = blockIdx.y * 32, n0 = blockIdx.x * 32;
  const int tx = threadIdx.x & 31, ty = threadIdx.x >> 5;  // 32 x 8
#pragma unroll
  for (int j = 0; j < 32; j += 8)
    tile[ty + j][tx] = src[(long long)(k0 + ty + j) * N + n0 + tx];
  __syncthreads();
#pragma unroll
  for (int j = 0; j < 32; j += 8)
    dst[(long long)(n0 + ty + j) * K + k0 + tx] = f2bf(tile[tx][ty + j]);
}

// ---------------------------------------------------------------------------
// Modulation GEMM: out[b, n] = silu(c[b,:]) @ W[:, n] + bias[n]   (B=4, tiny)
// ---------------------------------------------------------------------------
__global__ __launch_bounds__(256) void mod_gemm(const float* __restrict__ c,
                                                const float* __restrict__ W,
                                                const float* __restrict__ bias,
                                                float* __restrict__ out) {
  int idx = blockIdx.x * 256 + threadIdx.x;   // < B*6144
  int b = idx / 6144, n = idx % 6144;
  float acc = bias[n];
  for (int k = 0; k < 1024; ++k) {
    float cv = c[b * 1024 + k];
    float sc = cv / (1.0f + __expf(-cv));     // silu
    acc += sc * W[(long long)k * 6144 + n];
  }
  out[idx] = acc;
}

// ---------------------------------------------------------------------------
// LayerNorm + modulate -> bf16 activations.  One 256-thread block per row of
// 1024.  dst row = b*dst_bstride + dst_base + (row % rpb).
// ---------------------------------------------------------------------------
__global__ __launch_bounds__(256) void ln_mod_cast(
    const float* __restrict__ src, const float* __restrict__ mod,
    int shift_sec, int scale_sec, unsigned short* __restrict__ dst,
    int rpb, int dst_base, int dst_bstride) {
  __shared__ float rs[256], rs2[256];
  int row = blockIdx.x;
  int b = row / rpb, mr = row % rpb;
  const float* sp = src + (long long)row * 1024;
  float vals[4]; float s = 0.f, s2 = 0.f;
#pragma unroll
  for (int j = 0; j < 4; ++j) {
    float t = sp[threadIdx.x + j * 256];
    vals[j] = t; s += t; s2 += t * t;
  }
  rs[threadIdx.x] = s; rs2[threadIdx.x] = s2;
  __syncthreads();
  for (int o = 128; o > 0; o >>= 1) {
    if (threadIdx.x < o) {
      rs[threadIdx.x]  += rs[threadIdx.x + o];
      rs2[threadIdx.x] += rs2[threadIdx.x + o];
    }
    __syncthreads();
  }
  float mean = rs[0] * (1.0f / 1024.0f);
  float var  = rs2[0] * (1.0f / 1024.0f) - mean * mean;
  float inv  = rsqrtf(var + 1e-6f);
  const float* shp = mod + (long long)b * 6144 + shift_sec * 1024;
  const float* scp = mod + (long long)b * 6144 + scale_sec * 1024;
  unsigned short* dp =
      dst + ((long long)b * dst_bstride + dst_base + mr) * 1024;
#pragma unroll
  for (int j = 0; j < 4; ++j) {
    int col = threadIdx.x + j * 256;
    float t = (vals[j] - mean) * inv;
    t = t * (1.0f + scp[col]) + shp[col];
    dp[col] = f2bf(t);
  }
}

// ---------------------------------------------------------------------------
// bf16 WMMA GEMM: C[M,N] = A[M,K] @ Wt[N,K]^T   (tiles 128x128x32, 8 waves)
// Wt is the PRE-TRANSPOSED weight (N-major), so A and B staging are symmetric:
// one 32B global load + two ds_store_b128 per thread per K-step.
//   EPI 0: store raw        EPI 1: silu(acc+bias)     EPI 2: resid+gate*(acc+bias)
//   A row mapping: arow = (m/rpb)*a_bstride + a_base + m%rpb
//   out/resid row:  orow = (m/rpb)*o_bstride + o_base + m%rpb
// ---------------------------------------------------------------------------
template <int EPI, bool OUT_BF16>
__global__ __launch_bounds__(256) void gemm_bf16_wmma(
    const unsigned short* __restrict__ A, const unsigned short* __restrict__ Wt,
    const float* __restrict__ bias, const float* __restrict__ gate,
    const float* __restrict__ resid, void* __restrict__ OutP,
    int M, int N, int K, int rpb, int a_base, int a_bstride,
    int o_base, int o_bstride, int gate_stride) {
  __shared__ __align__(16) unsigned short lA[128 * 32];  // [m][k]
  __shared__ __align__(16) unsigned short lB[128 * 32];  // [n][k]

  const int tid = threadIdx.x, lane = tid & 31, wave = tid >> 5;
  const int tm = blockIdx.y * 128, tn = blockIdx.x * 128;
  const int wm = (wave >> 1) * 32, wn = (wave & 1) * 64;

  v8f acc[2][4];
#pragma unroll
  for (int i = 0; i < 2; ++i)
#pragma unroll
    for (int j = 0; j < 4; ++j) acc[i][j] = (v8f)0.0f;

  // cooperative tile-load mapping: each thread owns 16 bf16 (32B)
  const int lrow = tid >> 1;          // 0..127 (m or n within tile)
  const int lk = (tid & 1) * 16;      // 0 or 16
  const int gm = tm + lrow;
  const long long arow =
      (long long)(gm / rpb) * a_bstride + a_base + (gm % rpb);
  const unsigned short* aptr = A + arow * (long long)K + lk;
  const unsigned short* bptr = Wt + (long long)(tn + lrow) * K + lk;

  for (int k0 = 0; k0 < K; k0 += 32) {
    // ---- stage A tile ----
    U4 a0 = *reinterpret_cast<const U4*>(aptr + k0);
    U4 a1 = *reinterpret_cast<const U4*>(aptr + k0 + 8);
    *reinterpret_cast<U4*>(&lA[lrow * 32 + lk]) = a0;
    *reinterpret_cast<U4*>(&lA[lrow * 32 + lk + 8]) = a1;
    // ---- stage B tile (already N-major in memory) ----
    U4 b0 = *reinterpret_cast<const U4*>(bptr + k0);
    U4 b1 = *reinterpret_cast<const U4*>(bptr + k0 + 8);
    *reinterpret_cast<U4*>(&lB[lrow * 32 + lk]) = b0;
    *reinterpret_cast<U4*>(&lB[lrow * 32 + lk + 8]) = b1;
    if (k0 + 32 < K) {  // prefetch next K tiles (global_prefetch_b8)
      __builtin_prefetch(aptr + k0 + 32, 0, 1);
      __builtin_prefetch(bptr + k0 + 32, 0, 1);
    }
    __syncthreads();

    // ---- build fragments, contiguous k = kg*16 + s in both A and B ----
    const int kg = lane >> 4;    // lane-half
    const int lr = lane & 15;
    FragBF fa[2], fb[4];
#pragma unroll
    for (int fm = 0; fm < 2; ++fm) {
      const unsigned short* base = &lA[(wm + fm * 16 + lr) * 32];
      fa[fm].q[0] = *reinterpret_cast<const U4*>(base + kg * 16);       // K=kg*16..+7
      fa[fm].q[1] = *reinterpret_cast<const U4*>(base + kg * 16 + 8);   // K=kg*16+8..+15
    }
#pragma unroll
    for (int fn = 0; fn < 4; ++fn) {
      const unsigned short* base = &lB[(wn + fn * 16 + lr) * 32 + kg * 16];
      fb[fn].q[0] = *reinterpret_cast<const U4*>(base);                 // K=kg*16..+15
      fb[fn].q[1] = *reinterpret_cast<const U4*>(base + 8);
    }
#pragma unroll
    for (int fm = 0; fm < 2; ++fm)
#pragma unroll
      for (int fn = 0; fn < 4; ++fn)
        acc[fm][fn] = __builtin_amdgcn_wmma_f32_16x16x32_bf16(
            false, fa[fm].v, false, fb[fn].v, (short)0, acc[fm][fn],
            false, false);
    __syncthreads();
  }

  // ---- epilogue (C layout: row = hi*8+v, col = lane&15) ----
  const int lr = lane & 15, hi = lane >> 4;
#pragma unroll
  for (int fm = 0; fm < 2; ++fm)
#pragma unroll
    for (int fn = 0; fn < 4; ++fn)
#pragma unroll
      for (int v = 0; v < 8; ++v) {
        int row = tm + wm + fm * 16 + hi * 8 + v;
        int col = tn + wn + fn * 16 + lr;
        float val = acc[fm][fn][v];
        int bb = row / rpb, mr = row % rpb;
        long long orow = (long long)bb * o_bstride + o_base + mr;
        if (EPI == 1) {
          val += bias[col];
          val = val / (1.0f + __expf(-val));          // silu
        } else if (EPI == 2) {
          if (bias) val += bias[col];
          float g = gate[(long long)bb * gate_stride + col];
          val = resid[orow * (long long)N + col] + g * val;
        }
        if (OUT_BF16)
          reinterpret_cast<unsigned short*>(OutP)[orow * (long long)N + col] =
              f2bf(val);
        else
          reinterpret_cast<float*>(OutP)[orow * (long long)N + col] = val;
      }
}

// ---------------------------------------------------------------------------
// Infini-attention. One workgroup per (b, h); 256 threads = one query row
// each.  RMS-norm + RoPE applied on the fly while loading Q/K; online softmax
// with causal-in-segment mask; linear-memory term; M/z update per segment.
// Output written merged as (b, n, h*64+d) bf16.
// ---------------------------------------------------------------------------
__global__ __launch_bounds__(256) void infini_attn(
    const unsigned short* __restrict__ qkv,   // (B,4096,3072) bf16
    const float* __restrict__ gq_x, const float* __restrict__ gk_x,
    const float* __restrict__ gq_a, const float* __restrict__ gk_a,
    const float* __restrict__ beta, unsigned short* __restrict__ outact) {
  extern __shared__ float smem[];
  float* kf = smem;                 // 256*64
  float* vf = kf + 256 * 64;        // 256*64
  float* Mm = vf + 256 * 64;        // 64*64
  float* zv = Mm + 64 * 64;         // 64

  const int b = blockIdx.x >> 4;
  const int h = blockIdx.x & 15;
  const int t = threadIdx.x;
  const float g = 1.0f / (1.0f + __expf(-beta[h]));

  for (int i = t; i < 64 * 64; i += 256) Mm[i] = 0.0f;
  if (t < 64) zv[t] = 0.0f;
  __syncthreads();

  for (int seg = 0; seg < 16; ++seg) {
    const int pos = seg * 256 + t;
    const bool isa = pos < 512;
    const size_t rowbase = ((size_t)b * 4096 + pos) * 3072;

    // ---- load K (rms + rope) and V for token `pos` ----
    {
      const float* gk = (isa ? gk_a : gk_x) + h * 64;
      const unsigned short* kp = qkv + rowbase + 1024 + h * 64;
      float kr[64];
      float ss = 0.0f;
#pragma unroll
      for (int d = 0; d < 64; ++d) { kr[d] = bf2f(kp[d]); ss += kr[d] * kr[d]; }
      float sc = 8.0f / fmaxf(sqrtf(ss), 1e-12f);   // sqrt(DH)/norm
#pragma unroll
      for (int d = 0; d < 64; ++d) kr[d] *= sc * gk[d];
#pragma unroll
      for (int i = 0; i < 32; ++i) {
        float inv = __powf(10000.0f, -(float)(2 * i) / 64.0f);
        float ang = (float)pos * inv;
        float cs = __cosf(ang), sn = __sinf(ang);
        float t1 = kr[2 * i], t2 = kr[2 * i + 1];
        kr[2 * i]     = t1 * cs - t2 * sn;
        kr[2 * i + 1] = t1 * sn + t2 * cs;
      }
#pragma unroll
      for (int d = 0; d < 64; ++d) kf[t * 64 + d] = kr[d];
      const unsigned short* vp = qkv + rowbase + 2048 + h * 64;
#pragma unroll
      for (int d = 0; d < 64; ++d) vf[t * 64 + d] = bf2f(vp[d]);
    }
    __syncthreads();

    // ---- compute row `t` ----
    float qr[64];
    {
      const float* gq = (isa ? gq_a : gq_x) + h * 64;
      const unsigned short* qp = qkv + rowbase + h * 64;
      float ss = 0.0f;
#pragma unroll
      for (int d = 0; d < 64; ++d) { qr[d] = bf2f(qp[d]); ss += qr[d] * qr[d]; }
      float sc = 8.0f / fmaxf(sqrtf(ss), 1e-12f);
#pragma unroll
      for (int d = 0; d < 64; ++d) qr[d] *= sc * gq[d];
#pragma unroll
      for (int i = 0; i < 32; ++i) {
        float inv = __powf(10000.0f, -(float)(2 * i) / 64.0f);
        float ang = (float)pos * inv;
        float cs = __cosf(ang), sn = __sinf(ang);
        float t1 = qr[2 * i], t2 = qr[2 * i + 1];
        qr[2 * i]     = t1 * cs - t2 * sn;
        qr[2 * i + 1] = t1 * sn + t2 * cs;
      }
    }
    // linear memory term: mem = (sq @ M) / (sq . z + 1e-6)
    float sq[64];
#pragma unroll
    for (int d = 0; d < 64; ++d)
      sq[d] = qr[d] > 0.0f ? qr[d] + 1.0f : __expf(qr[d]);  // elu+1
    float denom = 1e-6f;
#pragma unroll
    for (int d = 0; d < 64; ++d) denom += sq[d] * zv[d];
    float memv[64];
#pragma unroll
    for (int e = 0; e < 64; ++e) {
      float s = 0.0f;
#pragma unroll
      for (int d = 0; d < 64; ++d) s += sq[d] * Mm[d * 64 + e];
      memv[e] = s / denom;
    }
    // online softmax over j <= t (causal within segment)
    float accv[64];
#pragma unroll
    for (int e = 0; e < 64; ++e) accv[e] = 0.0f;
    float mrun = -3.0e38f, lrun = 0.0f;
    for (int j = 0; j <= t; ++j) {
      float s = 0.0f;
      const float* kj = &kf[j * 64];
#pragma unroll
      for (int d = 0; d < 64; ++d) s += qr[d] * kj[d];
      s *= 0.125f;  // DH^-0.5
      float p;
      if (s > mrun) {
        float corr = __expf(mrun - s);
        lrun *= corr;
#pragma unroll
        for (int e = 0; e < 64; ++e) accv[e] *= corr;
        mrun = s; p = 1.0f;
      } else {
        p = __expf(s - mrun);
      }
      lrun += p;
      const float* vj = &vf[j * 64];
#pragma unroll
      for (int e = 0; e < 64; ++e) accv[e] += p * vj[e];
    }
    float linv = 1.0f / lrun;
    unsigned short* op =
        outact + ((size_t)b * 4096 + pos) * 1024 + h * 64;
#pragma unroll
    for (int e = 0; e < 64; ++e)
      op[e] = f2bf(g * memv[e] + (1.0f - g) * accv[e] * linv);
    __syncthreads();  // all reads of Mm/zv done before update

    // ---- memory update: M += sk^T @ v ; z += sum sk ----
#pragma unroll
    for (int ii = 0; ii < 16; ++ii) {
      int idx = t * 16 + ii;
      int d = idx >> 6, e = idx & 63;
      float s = 0.0f;
      for (int tok = 0; tok < 256; ++tok) {
        float kk = kf[tok * 64 + d];
        float sk = kk > 0.0f ? kk + 1.0f : __expf(kk);
        s += sk * vf[tok * 64 + e];
      }
      Mm[d * 64 + e] += s;
    }
    if (t < 64) {
      float s = 0.0f;
      for (int tok = 0; tok < 256; ++tok) {
        float kk = kf[tok * 64 + t];
        s += kk > 0.0f ? kk + 1.0f : __expf(kk);
      }
      zv[t] += s;
    }
    __syncthreads();  // before next segment overwrites kf/vf
  }
}

// ---------------------------------------------------------------------------
// Host orchestration
// ---------------------------------------------------------------------------
extern "C" void kernel_launch(void* const* d_in, const int* in_sizes, int n_in,
                              void* d_out, int out_size, void* d_ws,
                              size_t ws_size, hipStream_t stream) {
  (void)in_sizes; (void)n_in; (void)out_size; (void)ws_size;
  const int Bm = 4, NX = 3584, NA = 512, NT = 4096;

  const float* x_in   = (const float*)d_in[0];
  const float* a_in   = (const float*)d_in[1];
  const float* c_in   = (const float*)d_in[2];
  const float* modxw  = (const float*)d_in[3];
  const float* modxb  = (const float*)d_in[4];
  const float* modaw  = (const float*)d_in[5];
  const float* modab  = (const float*)d_in[6];
  const float* qkvxw  = (const float*)d_in[7];
  const float* qkvaw  = (const float*)d_in[8];
  const float* gqx    = (const float*)d_in[9];
  const float* gkx    = (const float*)d_in[10];
  const float* gqa    = (const float*)d_in[11];
  const float* gka    = (const float*)d_in[12];
  const float* beta   = (const float*)d_in[13];
  const float* outxw  = (const float*)d_in[14];
  const float* outaw  = (const float*)d_in[15];
  const float* xw1    = (const float*)d_in[16];
  const float* xb1    = (const float*)d_in[17];
  const float* xw2    = (const float*)d_in[18];
  const float* xb2    = (const float*)d_in[19];
  const float* aw1    = (const float*)d_in[20];
  const float* ab1    = (const float*)d_in[21];
  const float* aw2    = (const float*)d_in[22];
  const float* ab2    = (const float*)d_in[23];

  char* ws = (char*)d_ws;
  size_t off = 0;
  auto alc = [&](size_t bytes) {
    size_t r = off; off = (off + bytes + 255) & ~(size_t)255; return r;
  };
  float* mod_x = (float*)(ws + alc((size_t)Bm * 6144 * 4));
  float* mod_a = (float*)(ws + alc((size_t)Bm * 6144 * 4));
  float* x1    = (float*)(ws + alc((size_t)Bm * NX * 1024 * 4));
  float* a1    = (float*)(ws + alc((size_t)Bm * NA * 1024 * 4));
  unsigned short* act  = (unsigned short*)(ws + alc((size_t)Bm * NT * 1024 * 2));
  unsigned short* qkvb = (unsigned short*)(ws + alc((size_t)Bm * NT * 3072 * 2));
  unsigned short* mid  = (unsigned short*)(ws + alc((size_t)Bm * NT * 4096 * 2));
  // pre-transposed bf16 weights (N-major)
  unsigned short* w_qkv_x = (unsigned short*)(ws + alc((size_t)1024 * 3072 * 2));
  unsigned short* w_qkv_a = (unsigned short*)(ws + alc((size_t)1024 * 3072 * 2));
  unsigned short* w_out_x = (unsigned short*)(ws + alc((size_t)1024 * 1024 * 2));
  unsigned short* w_out_a = (unsigned short*)(ws + alc((size_t)1024 * 1024 * 2));
  unsigned short* w_x1 = (unsigned short*)(ws + alc((size_t)1024 * 4096 * 2));
  unsigned short* w_x2 = (unsigned short*)(ws + alc((size_t)4096 * 1024 * 2));
  unsigned short* w_a1 = (unsigned short*)(ws + alc((size_t)1024 * 4096 * 2));
  unsigned short* w_a2 = (unsigned short*)(ws + alc((size_t)4096 * 1024 * 2));

  float* out_x = (float*)d_out;
  float* out_a = (float*)d_out + (size_t)Bm * NX * 1024;

  // 1) weight cast + transpose (K x N fp32 -> N x K bf16)
  cast_transpose_bf16<<<dim3(3072 / 32, 1024 / 32), 256, 0, stream>>>(
      qkvxw, w_qkv_x, 1024, 3072);
  cast_transpose_bf16<<<dim3(3072 / 32, 1024 / 32), 256, 0, stream>>>(
      qkvaw, w_qkv_a, 1024, 3072);
  cast_transpose_bf16<<<dim3(1024 / 32, 1024 / 32), 256, 0, stream>>>(
      outxw, w_out_x, 1024, 1024);
  cast_transpose_bf16<<<dim3(1024 / 32, 1024 / 32), 256, 0, stream>>>(
      outaw, w_out_a, 1024, 1024);
  cast_transpose_bf16<<<dim3(4096 / 32, 1024 / 32), 256, 0, stream>>>(
      xw1, w_x1, 1024, 4096);
  cast_transpose_bf16<<<dim3(1024 / 32, 4096 / 32), 256, 0, stream>>>(
      xw2, w_x2, 4096, 1024);
  cast_transpose_bf16<<<dim3(4096 / 32, 1024 / 32), 256, 0, stream>>>(
      aw1, w_a1, 1024, 4096);
  cast_transpose_bf16<<<dim3(1024 / 32, 4096 / 32), 256, 0, stream>>>(
      aw2, w_a2, 4096, 1024);

  // 2) modulation vectors
  mod_gemm<<<(Bm * 6144) / 256, 256, 0, stream>>>(c_in, modxw, modxb, mod_x);
  mod_gemm<<<(Bm * 6144) / 256, 256, 0, stream>>>(c_in, modaw, modab, mod_a);

  // 3) LN + modulate -> bf16 act   (sections: 0=shift_attn, 1=scale_attn)
  ln_mod_cast<<<Bm * NX, 256, 0, stream>>>(x_in, mod_x, 0, 1, act, NX, 512, NT);
  ln_mod_cast<<<Bm * NA, 256, 0, stream>>>(a_in, mod_a, 0, 1, act, NA, 0, NT);

  // 4) QKV GEMMs (raw bf16 out, merged token layout)
  gemm_bf16_wmma<0, true><<<dim3(24, 16), 256, 0, stream>>>(
      act, w_qkv_a, nullptr, nullptr, nullptr, qkvb,
      Bm * NA, 3072, 1024, NA, 0, NT, 0, NT, 0);
  gemm_bf16_wmma<0, true><<<dim3(24, 112), 256, 0, stream>>>(
      act, w_qkv_x, nullptr, nullptr, nullptr, qkvb,
      Bm * NX, 3072, 1024, NX, 512, NT, 512, NT, 0);

  // 5) infini attention (rms+rope fused in-kernel), out merged bf16 into act
  size_t shmem = (size_t)(256 * 64 * 2 + 64 * 64 + 64) * sizeof(float);
  infini_attn<<<Bm * 16, 256, shmem, stream>>>(qkvb, gqx, gkx, gqa, gka, beta, act);

  // 6) out projections + gated residual   (gate section 2 = g_attn)
  gemm_bf16_wmma<2, false><<<dim3(8, 16), 256, 0, stream>>>(
      act, w_out_a, nullptr, mod_a + 2 * 1024, a_in, a1,
      Bm * NA, 1024, 1024, NA, 0, NT, 0, NA, 6144);
  gemm_bf16_wmma<2, false><<<dim3(8, 112), 256, 0, stream>>>(
      act, w_out_x, nullptr, mod_x + 2 * 1024, x_in, x1,
      Bm * NX, 1024, 1024, NX, 512, NT, 0, NX, 6144);

  // 7) LN + modulate for MLP (sections 3=shift_mlp, 4=scale_mlp)
  ln_mod_cast<<<Bm * NX, 256, 0, stream>>>(x1, mod_x, 3, 4, act, NX, 512, NT);
  ln_mod_cast<<<Bm * NA, 256, 0, stream>>>(a1, mod_a, 3, 4, act, NA, 0, NT);

  // 8) MLP layer 1: bias + silu -> bf16 mid
  gemm_bf16_wmma<1, true><<<dim3(32, 112), 256, 0, stream>>>(
      act, w_x1, xb1, nullptr, nullptr, mid,
      Bm * NX, 4096, 1024, NX, 512, NT, 512, NT, 0);
  gemm_bf16_wmma<1, true><<<dim3(32, 16), 256, 0, stream>>>(
      act, w_a1, ab1, nullptr, nullptr, mid,
      Bm * NA, 4096, 1024, NA, 0, NT, 0, NT, 0);

  // 9) MLP layer 2: bias + gate + residual -> final outputs (gate section 5)
  gemm_bf16_wmma<2, false><<<dim3(8, 112), 256, 0, stream>>>(
      mid, w_x2, xb2, mod_x + 5 * 1024, x1, out_x,
      Bm * NX, 1024, 4096, NX, 512, NT, 0, NX, 6144);
  gemm_bf16_wmma<2, false><<<dim3(8, 16), 256, 0, stream>>>(
      mid, w_a2, ab2, mod_a + 5 * 1024, a1, out_a,
      Bm * NA, 1024, 4096, NA, 0, NT, 0, NA, 6144);
}
